// multihead_self_attention_71098888618018
// MI455X (gfx1250) — matrix operand order repaired and checked by reading, not compiled, not run on hardware
//
#include <hip/hip_runtime.h>
#include <hip/hip_bf16.h>

// ---------------------------------------------------------------------------
// MHA forward with RoPE for MI455X (gfx1250), wave32 + v_wmma_f32_16x16x32_f16.
// B=2, S=2048, D_MODEL=1024, H=16, DK=64.
// GEMMs: 8-wave workgroups; weight tiles staged LDS via async global->LDS
//        (GLOBAL_LOAD_ASYNC_TO_LDS_B128, ASYNCcnt), manually 2x-unrolled
//        double buffering with fixed buffer roles (no accumulator copies).
// Attention: per-wave flash attention, software-pipelined K/V fragment loads.
// ---------------------------------------------------------------------------

typedef _Float16 v16h __attribute__((ext_vector_type(16)));
typedef _Float16 v8h  __attribute__((ext_vector_type(8)));
typedef float    v8f  __attribute__((ext_vector_type(8)));
typedef float    f4v  __attribute__((ext_vector_type(4)));

#define D_MODEL 1024
#define NHEADS  16
#define DK      64
#define SEQ     2048
#define BATCH   2
#define MROWS   (BATCH * SEQ)            // 4096 token rows
#define LN_THETA 9.210340371976184f      // ln(10000)

union H16 { v16h v; v8h h[2]; };

// A-fragment (16x32 f16, M x K): lane m = lane&15; lanes<16 hold K = 0..15,
// lanes>=16 hold K = 16..31 (kb = 0 / 8, so start = 2*kb).
__device__ __forceinline__ v16h load_afrag(const _Float16* rowPtr, int kb) {
  H16 a;
  a.h[0] = *(const v8h*)(rowPtr + 2 * kb);
  a.h[1] = *(const v8h*)(rowPtr + 2 * kb + 8);
  return a.v;
}

__device__ __forceinline__ v8f wmma_f16(v16h a, v16h b, v8f c) {
  return __builtin_amdgcn_wmma_f32_16x16x32_f16(false, a, false, b,
                                                (short)0, c, false, false);
}

// Async global->LDS copy (16 B per lane), tracked with ASYNCcnt.
// ldsAddr is the LDS byte offset: flat shared-aperture addresses carry the
// LDS offset in addr[31:0] (ISA 10.2), which is what VDST expects.
__device__ __forceinline__ void async_ld_b128(unsigned ldsAddr, const void* gptr) {
  unsigned long long ga = (unsigned long long)gptr;
  asm volatile("global_load_async_to_lds_b128 %0, %1, off"
               :: "v"(ldsAddr), "v"(ga) : "memory");
}
__device__ __forceinline__ void wait_async0() {
  asm volatile("s_wait_asynccnt 0x0" ::: "memory");
}

// ---------------------------------------------------------------------------
// fp32 -> fp16 conversion, 8 elements per thread
// ---------------------------------------------------------------------------
__global__ __launch_bounds__(256) void f32_to_f16_kernel(const float* __restrict__ in,
                                                         _Float16* __restrict__ out,
                                                         int n8) {
  int i = blockIdx.x * 256 + threadIdx.x;
  if (i < n8) {
    const f4v* p = (const f4v*)in;
    f4v a = p[2 * i];
    f4v b = p[2 * i + 1];
    v8h o;
    o[0] = (_Float16)a[0]; o[1] = (_Float16)a[1];
    o[2] = (_Float16)a[2]; o[3] = (_Float16)a[3];
    o[4] = (_Float16)b[0]; o[5] = (_Float16)b[1];
    o[6] = (_Float16)b[2]; o[7] = (_Float16)b[3];
    ((v8h*)out)[i] = o;
  }
}

// ---------------------------------------------------------------------------
// Fused QKV projection (WMMA, async-LDS-staged weights) + RoPE.
//   grid = (16 heads, 32 super-tiles of 128 rows, 3 {Q,K,V}), block = 256.
//   Weight chunk 64(N) x 32(K) f16 staged via async copy, double buffered.
// ---------------------------------------------------------------------------
__global__ __launch_bounds__(256) void qkv_rope_kernel(
    const _Float16* __restrict__ x_h,
    const _Float16* __restrict__ Wq_h,
    const _Float16* __restrict__ Wk_h,
    const _Float16* __restrict__ Wv_h,
    const int* __restrict__ tokpos,
    _Float16* __restrict__ Q_h,
    _Float16* __restrict__ K_h,
    _Float16* __restrict__ Vt_h) {
  __shared__ __align__(32) _Float16 Wl[2][64 * 32];

  const int tid  = threadIdx.x;
  const int lane = tid & 31;
  const int wave = tid >> 5;             // 0..7
  const int head = blockIdx.x;           // 0..15
  const int tm   = blockIdx.y;           // 0..31
  const int z    = blockIdx.z;           // 0=Q 1=K 2=V (uniform)
  const _Float16* W = (z == 0) ? Wq_h : ((z == 1) ? Wk_h : Wv_h);

  const int nn     = lane & 15;
  const int grp    = lane >> 4;
  const int kbA    = grp * 8;
  const int kbB    = grp * 16;
  const int rowoff = grp * 8;
  const int m0     = tm * 128 + wave * 16;
  const int nbase  = head * 64;

  // cooperative W staging: thread tid moves 16 B per k-chunk.
  const int li = tid * 8;                // 0..2047
  const int wn = li >> 5;                // n_local 0..63
  const int wk = li & 31;                // k_local in {0,8,16,24}
  const _Float16* Wg = W + (size_t)(nbase + wn) * D_MODEL + wk;
  const unsigned lds0 = (unsigned)(size_t)(&Wl[0][wn * 32 + wk]);
  const unsigned lds1 = (unsigned)(size_t)(&Wl[1][wn * 32 + wk]);

  const _Float16* Arow = x_h + (size_t)(m0 + nn) * D_MODEL;

  // prologue: stage chunk 0 -> Wl[0]; A fragment for chunk 0
  async_ld_b128(lds0, Wg);
  v16h a0 = load_afrag(Arow, kbA);
  wait_async0();
  __syncthreads();

  v8f cc[4] = {};
  int k = 0;
#pragma unroll 1
  for (int i = 0; i < 15; ++i, k += 64) {
    // --- half 1: compute chunk k from Wl[0]; stage chunk k+32 -> Wl[1] ---
    async_ld_b128(lds1, Wg + k + 32);
    __builtin_prefetch(Arow + k + 64, 0, 3);
    v16h a1 = load_afrag(Arow + k + 32, kbA);
#pragma unroll
    for (int t = 0; t < 4; ++t) {
      v16h b = *(const v16h*)(&Wl[0][(t * 16 + nn) * 32 + kbB]);
      cc[t] = wmma_f16(a0, b, cc[t]);
    }
    wait_async0();
    __syncthreads();
    // --- half 2: compute chunk k+32 from Wl[1]; stage chunk k+64 -> Wl[0] ---
    async_ld_b128(lds0, Wg + k + 64);
    a0 = load_afrag(Arow + k + 64, kbA);
#pragma unroll
    for (int t = 0; t < 4; ++t) {
      v16h b = *(const v16h*)(&Wl[1][(t * 16 + nn) * 32 + kbB]);
      cc[t] = wmma_f16(a1, b, cc[t]);
    }
    wait_async0();
    __syncthreads();
  }
  {   // epilogue: k == 960; Wl[0] holds chunk 960, a0 = A(960)
    async_ld_b128(lds1, Wg + 992);
    v16h a1 = load_afrag(Arow + 992, kbA);
#pragma unroll
    for (int t = 0; t < 4; ++t) {
      v16h b = *(const v16h*)(&Wl[0][(t * 16 + nn) * 32 + kbB]);
      cc[t] = wmma_f16(a0, b, cc[t]);
    }
    wait_async0();
    __syncthreads();
#pragma unroll
    for (int t = 0; t < 4; ++t) {
      v16h b = *(const v16h*)(&Wl[1][(t * 16 + nn) * 32 + kbB]);
      cc[t] = wmma_f16(a1, b, cc[t]);
    }
  }

  // ---- RoPE + store ----
#pragma unroll
  for (int r = 0; r < 8; ++r) {
    const int mg = m0 + r + rowoff;          // global token row
    const int bb = mg >> 11;                 // / SEQ
    const int ss = mg & (SEQ - 1);           // % SEQ
    const float pos = (float)tokpos[mg];
#pragma unroll
    for (int t = 0; t < 4; ++t) {
      const int d = t * 16 + nn;             // column inside head (0..63)
      float val = cc[t][r];
      if (z < 2) {
        // RoPE pair (2i, 2i+1) sits in adjacent lanes of the C layout.
        float partner = __shfl_xor(val, 1, 32);
        const float freq = __expf(-LN_THETA * (float)(d & ~1) * (1.0f / 64.0f));
        float sv, cv;
        __sincosf(pos * freq, &sv, &cv);
        val = ((d & 1) == 0) ? (val * cv - partner * sv)
                             : (val * cv + partner * sv);
        if (z == 0) val *= 0.125f;           // 1/sqrt(64) folded into Q
        _Float16* dst = (z == 0) ? Q_h : K_h;
        dst[((size_t)(bb * NHEADS + head) * SEQ + ss) * DK + d] = (_Float16)val;
      } else {
        Vt_h[((size_t)(bb * NHEADS + head) * DK + d) * SEQ + ss] = (_Float16)val;
      }
    }
  }
}

// ---------------------------------------------------------------------------
// Flash attention: one wave handles 16 query rows of one (b,h).
//   grid = (S/16 = 128, B*H = 32), block = 32.
// ---------------------------------------------------------------------------
__global__ __launch_bounds__(32) void attn_kernel(
    const _Float16* __restrict__ Q_h,
    const _Float16* __restrict__ K_h,
    const _Float16* __restrict__ Vt_h,
    _Float16* __restrict__ att_h) {
  __shared__ __align__(32) _Float16 Pl[16 * 32];

  const int lane = threadIdx.x;
  const int qt   = blockIdx.x;           // 0..127
  const int bh   = blockIdx.y;           // 0..31
  const int b    = bh >> 4;
  const int h    = bh & 15;
  const int q0   = qt * 16;

  const int nn     = lane & 15;
  const int grp    = lane >> 4;
  const int kbA    = grp * 8;
  const int kbB    = grp * 16;
  const int rowoff = grp * 8;

  // Q fragments (Q already carries the 1/8 scale from the projection).
  const _Float16* Qrow = Q_h + ((size_t)bh * SEQ + q0 + nn) * DK;
  v16h aQ0 = load_afrag(Qrow, kbA);
  v16h aQ1 = load_afrag(Qrow + 32, kbA);

  v8f o0 = {}, o1 = {}, o2 = {}, o3 = {};
  float mrow[8], lrow[8];
#pragma unroll
  for (int r = 0; r < 8; ++r) { mrow[r] = -1e30f; lrow[r] = 0.0f; }

  const _Float16* Kbase = K_h  + (size_t)bh * SEQ * DK;
  const _Float16* Vbase = Vt_h + (size_t)bh * DK * SEQ;

  // prologue: K fragments for chunk 0
  v16h b00 = *(const v16h*)(Kbase + (size_t)nn * DK + kbB);
  v16h b01 = *(const v16h*)(Kbase + (size_t)nn * DK + kbB + 32);
  v16h b10 = *(const v16h*)(Kbase + (size_t)(16 + nn) * DK + kbB);
  v16h b11 = *(const v16h*)(Kbase + (size_t)(16 + nn) * DK + kbB + 32);

  const int nchunks = qt / 2 + 1;        // causal: keys 0 .. q0+15
  for (int c = 0; c < nchunks; ++c) {
    const int k0 = c * 32;

    // V fragments for the current chunk (issued early; consumed after softmax)
    const _Float16* Vcol = Vbase + (size_t)nn * SEQ + k0 + kbB;
    v16h bv0 = *(const v16h*)(Vcol);
    v16h bv1 = *(const v16h*)(Vcol + 16 * SEQ);
    v16h bv2 = *(const v16h*)(Vcol + 32 * SEQ);
    v16h bv3 = *(const v16h*)(Vcol + 48 * SEQ);

    // ---- scores: S = Q K^T (K frags loaded in the previous iteration) ----
    v8f s0 = {}, s1 = {};
    s0 = wmma_f16(aQ0, b00, s0);
    s0 = wmma_f16(aQ1, b01, s0);
    s1 = wmma_f16(aQ0, b10, s1);
    s1 = wmma_f16(aQ1, b11, s1);

    // prefetch next chunk's K fragments (clamped reload on last iteration)
    const int kn = (c + 1 < nchunks) ? (k0 + 32) : k0;
    const _Float16* Knr0 = Kbase + (size_t)(kn + nn) * DK + kbB;
    const _Float16* Knr1 = Kbase + (size_t)(kn + 16 + nn) * DK + kbB;
    v16h nb00 = *(const v16h*)(Knr0);
    v16h nb01 = *(const v16h*)(Knr0 + 32);
    v16h nb10 = *(const v16h*)(Knr1);
    v16h nb11 = *(const v16h*)(Knr1 + 32);

    // ---- causal mask + online softmax (16-lane butterfly reductions) ----
#pragma unroll
    for (int r = 0; r < 8; ++r) {
      const int qrow = q0 + r + rowoff;
      float v0 = ((k0 + nn)      <= qrow) ? s0[r] : -1e30f;
      float v1 = ((k0 + 16 + nn) <= qrow) ? s1[r] : -1e30f;
      float sm = fmaxf(v0, v1);
#pragma unroll
      for (int off = 1; off < 16; off <<= 1)
        sm = fmaxf(sm, __shfl_xor(sm, off, 32));
      const float mnew  = fmaxf(mrow[r], sm);
      const float alpha = __expf(mrow[r] - mnew);
      const float p0 = __expf(v0 - mnew);
      const float p1 = __expf(v1 - mnew);
      float rs = p0 + p1;
#pragma unroll
      for (int off = 1; off < 16; off <<= 1)
        rs += __shfl_xor(rs, off, 32);
      lrow[r] = lrow[r] * alpha + rs;
      mrow[r] = mnew;
      o0[r] *= alpha; o1[r] *= alpha; o2[r] *= alpha; o3[r] *= alpha;
      const int prow = r + rowoff;
      Pl[prow * 32 + nn]      = (_Float16)p0;
      Pl[prow * 32 + 16 + nn] = (_Float16)p1;
    }
    __syncthreads();

    // ---- O += P(16x32) @ V(32x64): one A frag (via LDS), four B frags ----
    v16h aP = load_afrag(Pl + nn * 32, kbA);
    o0 = wmma_f16(aP, bv0, o0);
    o1 = wmma_f16(aP, bv1, o1);
    o2 = wmma_f16(aP, bv2, o2);
    o3 = wmma_f16(aP, bv3, o3);
    __syncthreads();

    b00 = nb00; b01 = nb01; b10 = nb10; b11 = nb11;
  }

  // ---- normalize and emit f16 attention output [b*S+s][h*64+d] ----
#pragma unroll
  for (int r = 0; r < 8; ++r) {
    const float inv = 1.0f / lrow[r];
    const int srow = q0 + r + rowoff;
    _Float16* dst = att_h + ((size_t)(b * SEQ) + srow) * D_MODEL + h * DK;
    dst[nn]      = (_Float16)(o0[r] * inv);
    dst[16 + nn] = (_Float16)(o1[r] * inv);
    dst[32 + nn] = (_Float16)(o2[r] * inv);
    dst[48 + nn] = (_Float16)(o3[r] * inv);
  }
}

// ---------------------------------------------------------------------------
// Output projection: out[4096x1024] (f32) = att (f16) @ Wo^T (f16).
//   Same async-staged, manually 2x-unrolled structure as the QKV GEMM.
//   grid = (16, 32), block = 256.
// ---------------------------------------------------------------------------
__global__ __launch_bounds__(256) void out_proj_kernel(
    const _Float16* __restrict__ att_h,
    const _Float16* __restrict__ Wo_h,
    float* __restrict__ out) {
  __shared__ __align__(32) _Float16 Wl[2][64 * 32];

  const int tid  = threadIdx.x;
  const int lane = tid & 31;
  const int wave = tid >> 5;
  const int tn   = blockIdx.x;           // 0..15
  const int tm   = blockIdx.y;           // 0..31
  const int nn     = lane & 15;
  const int grp    = lane >> 4;
  const int kbA    = grp * 8;
  const int kbB    = grp * 16;
  const int rowoff = grp * 8;
  const int m0     = tm * 128 + wave * 16;
  const int nbase  = tn * 64;

  const int li = tid * 8;
  const int wn = li >> 5;
  const int wk = li & 31;
  const _Float16* Wg = Wo_h + (size_t)(nbase + wn) * D_MODEL + wk;
  const unsigned lds0 = (unsigned)(size_t)(&Wl[0][wn * 32 + wk]);
  const unsigned lds1 = (unsigned)(size_t)(&Wl[1][wn * 32 + wk]);

  const _Float16* Arow = att_h + (size_t)(m0 + nn) * D_MODEL;

  async_ld_b128(lds0, Wg);
  v16h a0 = load_afrag(Arow, kbA);
  wait_async0();
  __syncthreads();

  v8f cc[4] = {};
  int k = 0;
#pragma unroll 1
  for (int i = 0; i < 15; ++i, k += 64) {
    async_ld_b128(lds1, Wg + k + 32);
    __builtin_prefetch(Arow + k + 64, 0, 3);
    v16h a1 = load_afrag(Arow + k + 32, kbA);
#pragma unroll
    for (int t = 0; t < 4; ++t) {
      v16h b = *(const v16h*)(&Wl[0][(t * 16 + nn) * 32 + kbB]);
      cc[t] = wmma_f16(a0, b, cc[t]);
    }
    wait_async0();
    __syncthreads();

    async_ld_b128(lds0, Wg + k + 64);
    a0 = load_afrag(Arow + k + 64, kbA);
#pragma unroll
    for (int t = 0; t < 4; ++t) {
      v16h b = *(const v16h*)(&Wl[1][(t * 16 + nn) * 32 + kbB]);
      cc[t] = wmma_f16(a1, b, cc[t]);
    }
    wait_async0();
    __syncthreads();
  }
  {
    async_ld_b128(lds1, Wg + 992);
    v16h a1 = load_afrag(Arow + 992, kbA);
#pragma unroll
    for (int t = 0; t < 4; ++t) {
      v16h b = *(const v16h*)(&Wl[0][(t * 16 + nn) * 32 + kbB]);
      cc[t] = wmma_f16(a0, b, cc[t]);
    }
    wait_async0();
    __syncthreads();
#pragma unroll
    for (int t = 0; t < 4; ++t) {
      v16h b = *(const v16h*)(&Wl[1][(t * 16 + nn) * 32 + kbB]);
      cc[t] = wmma_f16(a1, b, cc[t]);
    }
  }

#pragma unroll
  for (int t = 0; t < 4; ++t)
#pragma unroll
    for (int r = 0; r < 8; ++r)
      out[(size_t)(m0 + r + rowoff) * D_MODEL + nbase + t * 16 + nn] = cc[t][r];
}

// ---------------------------------------------------------------------------
// Host-side launch.  Inputs: x, W_q, W_k, W_v, W_o (f32), token_positions (i32).
// Workspace layout (48 MiB total):
//   [0,8)   x_h      [8,10) Wq_h  [10,12) Wk_h  [12,14) Wv_h  [14,16) Wo_h
//   [16,24) Q_h      [24,32) K_h  [32,40) Vt_h  [40,48) att_h      (MiB)
// ---------------------------------------------------------------------------
extern "C" void kernel_launch(void* const* d_in, const int* in_sizes, int n_in,
                              void* d_out, int out_size, void* d_ws, size_t ws_size,
                              hipStream_t stream) {
  (void)in_sizes; (void)n_in; (void)out_size; (void)ws_size;
  const float* x      = (const float*)d_in[0];
  const float* Wq     = (const float*)d_in[1];
  const float* Wk     = (const float*)d_in[2];
  const float* Wv     = (const float*)d_in[3];
  const float* Wo     = (const float*)d_in[4];
  const int*   tokpos = (const int*)d_in[5];
  float* out = (float*)d_out;

  char* ws = (char*)d_ws;
  const size_t MB = 1024u * 1024u;
  _Float16* x_h   = (_Float16*)(ws);
  _Float16* Wq_h  = (_Float16*)(ws + 8 * MB);
  _Float16* Wk_h  = (_Float16*)(ws + 10 * MB);
  _Float16* Wv_h  = (_Float16*)(ws + 12 * MB);
  _Float16* Wo_h  = (_Float16*)(ws + 14 * MB);
  _Float16* Q_h   = (_Float16*)(ws + 16 * MB);
  _Float16* K_h   = (_Float16*)(ws + 24 * MB);
  _Float16* Vt_h  = (_Float16*)(ws + 32 * MB);
  _Float16* att_h = (_Float16*)(ws + 40 * MB);

  const int nx8 = (MROWS * D_MODEL) / 8;     // 524,288
  const int nw8 = (D_MODEL * D_MODEL) / 8;   // 131,072
  f32_to_f16_kernel<<<(nx8 + 255) / 256, 256, 0, stream>>>(x,  x_h,  nx8);
  f32_to_f16_kernel<<<(nw8 + 255) / 256, 256, 0, stream>>>(Wq, Wq_h, nw8);
  f32_to_f16_kernel<<<(nw8 + 255) / 256, 256, 0, stream>>>(Wk, Wk_h, nw8);
  f32_to_f16_kernel<<<(nw8 + 255) / 256, 256, 0, stream>>>(Wv, Wv_h, nw8);
  f32_to_f16_kernel<<<(nw8 + 255) / 256, 256, 0, stream>>>(Wo, Wo_h, nw8);

  qkv_rope_kernel<<<dim3(NHEADS, MROWS / 128, 3), 256, 0, stream>>>(
      x_h, Wq_h, Wk_h, Wv_h, tokpos, Q_h, K_h, Vt_h);

  attn_kernel<<<dim3(SEQ / 16, BATCH * NHEADS), 32, 0, stream>>>(
      Q_h, K_h, Vt_h, att_h);

  out_proj_kernel<<<dim3(NHEADS, MROWS / 128), 256, 0, stream>>>(att_h, Wo_h, out);
}
